// QuantumProxyGNN_23510650978817
// MI455X (gfx1250) — matrix operand order repaired and checked by reading, not compiled, not run on hardware
//
#include <hip/hip_runtime.h>
#include <hip/hip_bf16.h>

typedef __attribute__((ext_vector_type(16))) _Float16 v16h;
typedef __attribute__((ext_vector_type(8)))  float    v8f;

#define N_NODES  50000
#define N_EDGES  800000
#define N_GRAPHS 512
#define F_IN     16
#define HID      128

// ---------------------------------------------------------------- utilities
__global__ __launch_bounds__(256) void k_fill(float* p, float v, int n) {
    int i = blockIdx.x * 256 + threadIdx.x;
    if (i < n) p[i] = v;
}

__global__ __launch_bounds__(256) void k_deg_count(const int* __restrict__ dst,
                                                   float* __restrict__ deg, int E) {
    int e = blockIdx.x * 256 + threadIdx.x;
    if (e < E) atomicAdd(&deg[dst[e]], 1.0f);
}

__global__ __launch_bounds__(256) void k_rsqrt(float* p, int n) {
    int i = blockIdx.x * 256 + threadIdx.x;
    if (i < n) p[i] = rsqrtf(p[i]);
}

// ---------------------------------------------------------------- WMMA GEMM
// C[M,128] = A[M,K] @ W[K,128], fp32 in/out, f16 WMMA with f32 accumulate.
// Block = 256 threads (8 waves); each block computes a 128x128 tile of C.
// W is staged transposed ([N][Kpad]) in LDS as f16 so each lane's B fragment
// is a contiguous 16-element K-run (2x ds_load_b128).
template <int K>
__global__ __launch_bounds__(256) void k_gemm_wmma(const float* __restrict__ A,
                                                   const float* __restrict__ W,
                                                   float* __restrict__ C, int M) {
    constexpr int KP = (K < 32) ? 32 : K;   // pad K to one 32-wide WMMA step
    __shared__ _Float16 Wt[128][KP];

    const int tid = threadIdx.x;
    // Load + transpose + convert W -> LDS (zeros in the K pad region).
    for (int i = tid; i < 128 * KP; i += 256) {
        int n = i / KP, k = i % KP;
        float w = (k < K) ? W[k * 128 + n] : 0.0f;
        Wt[n][k] = (_Float16)w;
    }
    __syncthreads();

    const int wave = tid >> 5;
    const int lane = tid & 31;
    const int m    = lane & 15;      // row-in-tile for A, col-in-tile for B/D
    const int hi   = lane >> 4;      // lane half selects K sub-range
    const int row0 = blockIdx.x * 128 + wave * 16;
    const int arow = row0 + m;
    const bool rv  = (arow < M);
    const float* ap = A + (size_t)(rv ? arow : 0) * K;

    v8f acc[8];
#pragma unroll
    for (int n = 0; n < 8; ++n) acc[n] = (v8f){0,0,0,0,0,0,0,0};

#pragma unroll
    for (int kk = 0; kk < KP; kk += 32) {
        if (kk + 32 < KP) __builtin_prefetch(ap + kk + 32, 0, 1);
        // A fragment 16x32 f16: elems 0..15 -> K=kk+hi*16+e (contiguous).
        v16h a;
#pragma unroll
        for (int e = 0; e < 8; ++e) {
            int k0 = kk + hi * 16 + e;
            int k1 = k0 + 8;
            float f0 = (rv && k0 < K) ? ap[k0] : 0.0f;
            float f1 = (rv && k1 < K) ? ap[k1] : 0.0f;
            a[e]     = (_Float16)f0;
            a[e + 8] = (_Float16)f1;
        }
#pragma unroll
        for (int n = 0; n < 8; ++n) {
            // B fragment 32x16: lane col = m, K = kk + hi*16 + e (contiguous).
            v16h b = *(const v16h*)&Wt[n * 16 + m][kk + hi * 16];
            acc[n] = __builtin_amdgcn_wmma_f32_16x16x32_f16(
                false, a, false, b, (short)0, acc[n], false, false);
        }
    }

    // D layout: VGPR r -> row = row0 + r + 8*hi, col = n*16 + m.
#pragma unroll
    for (int r = 0; r < 8; ++r) {
        int orow = row0 + r + 8 * hi;
        if (orow < M) {
            float* cp = C + (size_t)orow * 128 + m;
#pragma unroll
            for (int n = 0; n < 8; ++n) cp[n * 16] = acc[n][r];
        }
    }
}

// ------------------------------------------------------- edge scatter (GCN)
// One wave per edge: 32 lanes x float4 = 128 features. h fits in L2 (25.6MB).
__global__ __launch_bounds__(256) void k_edge_agg(const int* __restrict__ ei,
                                                  const float* __restrict__ h,
                                                  const float* __restrict__ dinv,
                                                  float* __restrict__ agg, int E) {
    int e = blockIdx.x * 8 + (threadIdx.x >> 5);
    if (e >= E) return;
    int lane = threadIdx.x & 31;
    int s = ei[e];
    int d = ei[E + e];
    float c = dinv[s] * dinv[d];
    float4 v = ((const float4*)(h + (size_t)s * 128))[lane];
    float* ap = agg + (size_t)d * 128 + lane * 4;
    atomicAdd(ap + 0, v.x * c);
    atomicAdd(ap + 1, v.y * c);
    atomicAdd(ap + 2, v.z * c);
    atomicAdd(ap + 3, v.w * c);
}

// agg += h * dinv^2 + bias ; ReLU ; in-place on agg
__global__ __launch_bounds__(256) void k_combine_relu(float* __restrict__ agg,
                                                      const float* __restrict__ h,
                                                      const float* __restrict__ dinv,
                                                      const float* __restrict__ bias,
                                                      int total) {
    int i = blockIdx.x * 256 + threadIdx.x;
    if (i >= total) return;
    int node = i >> 7, f = i & 127;
    float di = dinv[node];
    float v = agg[i] + h[i] * di * di + bias[f];
    agg[i] = fmaxf(v, 0.0f);
}

// ------------------------------------------------------------- mean pooling
__global__ __launch_bounds__(256) void k_pool_sum(const float* __restrict__ h,
                                                  const int* __restrict__ batch,
                                                  float* __restrict__ g,
                                                  float* __restrict__ cnt, int N) {
    int node = blockIdx.x * 8 + (threadIdx.x >> 5);
    if (node >= N) return;
    int lane = threadIdx.x & 31;
    int gi = batch[node];
    float4 v = ((const float4*)(h + (size_t)node * 128))[lane];
    float* gp = g + (size_t)gi * 128 + lane * 4;
    atomicAdd(gp + 0, v.x);
    atomicAdd(gp + 1, v.y);
    atomicAdd(gp + 2, v.z);
    atomicAdd(gp + 3, v.w);
    if (lane == 0) atomicAdd(&cnt[gi], 1.0f);
}

__global__ __launch_bounds__(256) void k_pool_div(float* __restrict__ g,
                                                  const float* __restrict__ cnt,
                                                  int total) {
    int i = blockIdx.x * 256 + threadIdx.x;
    if (i < total) g[i] /= fmaxf(cnt[i >> 7], 1.0f);
}

// ----------------------------------------------------------------- MLP heads
// One 64-thread block per graph; computes both heads.
__global__ __launch_bounds__(64) void k_heads(const float* __restrict__ g,
                                              const float* __restrict__ Wt1,
                                              const float* __restrict__ bt1,
                                              const float* __restrict__ Wt2,
                                              const float* __restrict__ bt2,
                                              const float* __restrict__ Wc1,
                                              const float* __restrict__ bc1,
                                              const float* __restrict__ Wc2,
                                              const float* __restrict__ bc2,
                                              float* __restrict__ out, int G) {
    int gi = blockIdx.x;
    int j = threadIdx.x;
    __shared__ float gs[128];
    __shared__ float red[64];
    gs[j]      = g[(size_t)gi * 128 + j];
    gs[j + 64] = g[(size_t)gi * 128 + 64 + j];
    __syncthreads();

    float ht = bt1[j];
    for (int k = 0; k < 128; ++k) ht = fmaf(gs[k], Wt1[k * 64 + j], ht);
    red[j] = fmaxf(ht, 0.0f) * Wt2[j];
    __syncthreads();
    for (int s = 32; s > 0; s >>= 1) {
        if (j < s) red[j] += red[j + s];
        __syncthreads();
    }
    if (j == 0) out[gi] = red[0] + bt2[0];
    __syncthreads();

    float hc = bc1[j];
    for (int k = 0; k < 128; ++k) hc = fmaf(gs[k], Wc1[k * 64 + j], hc);
    red[j] = fmaxf(hc, 0.0f) * Wc2[j];
    __syncthreads();
    for (int s = 32; s > 0; s >>= 1) {
        if (j < s) red[j] += red[j + s];
        __syncthreads();
    }
    if (j == 0) out[G + gi] = 1.0f / (1.0f + expf(-(red[0] + bc2[0])));
}

// ------------------------------------------------------------------ launcher
extern "C" void kernel_launch(void* const* d_in, const int* in_sizes, int n_in,
                              void* d_out, int out_size, void* d_ws, size_t ws_size,
                              hipStream_t stream) {
    const float* x     = (const float*)d_in[0];
    const int*   ei    = (const int*)d_in[1];     // [2, E]
    const int*   batch = (const int*)d_in[2];
    const float* W1 = (const float*)d_in[3];  const float* b1 = (const float*)d_in[4];
    const float* W2 = (const float*)d_in[5];  const float* b2 = (const float*)d_in[6];
    const float* W3 = (const float*)d_in[7];  const float* b3 = (const float*)d_in[8];
    const float* Wt1 = (const float*)d_in[9];  const float* bt1 = (const float*)d_in[10];
    const float* Wt2 = (const float*)d_in[11]; const float* bt2 = (const float*)d_in[12];
    const float* Wc1 = (const float*)d_in[13]; const float* bc1 = (const float*)d_in[14];
    const float* Wc2 = (const float*)d_in[15]; const float* bc2 = (const float*)d_in[16];
    float* out = (float*)d_out;

    const int N = N_NODES, E = N_EDGES, G = N_GRAPHS;
    const int NH = N * HID;

    // workspace layout (floats)
    float* ws   = (float*)d_ws;
    float* dinv = ws;                    // N
    float* B0   = ws + 50048;            // N*HID (gemm output h = A@W)
    float* B1   = B0 + NH;               // N*HID (agg target / layer output)
    float* g    = B1 + NH;               // G*HID
    float* cnt  = g + G * HID;           // G

    const int nb256 = (N + 255) / 256;
    const int eb256 = (E + 255) / 256;
    const int nhb   = (NH + 255) / 256;
    const int gemmb = (N + 127) / 128;
    const int edgeb = (E + 7) / 8;
    const int poolb = (N + 7) / 8;

    // degrees: deg = 1 + indegree(A); dinv = rsqrt(deg)
    k_fill<<<nb256, 256, 0, stream>>>(dinv, 1.0f, N);
    k_deg_count<<<eb256, 256, 0, stream>>>(ei + E, dinv, E);
    k_rsqrt<<<nb256, 256, 0, stream>>>(dinv, N);

    // ---- layer 1 (K = 16)
    k_gemm_wmma<F_IN><<<gemmb, 256, 0, stream>>>(x, W1, B0, N);
    k_fill<<<nhb, 256, 0, stream>>>(B1, 0.0f, NH);
    k_edge_agg<<<edgeb, 256, 0, stream>>>(ei, B0, dinv, B1, E);
    k_combine_relu<<<nhb, 256, 0, stream>>>(B1, B0, dinv, b1, NH);

    // ---- layer 2 (K = 128)
    k_gemm_wmma<HID><<<gemmb, 256, 0, stream>>>(B1, W2, B0, N);
    k_fill<<<nhb, 256, 0, stream>>>(B1, 0.0f, NH);
    k_edge_agg<<<edgeb, 256, 0, stream>>>(ei, B0, dinv, B1, E);
    k_combine_relu<<<nhb, 256, 0, stream>>>(B1, B0, dinv, b2, NH);

    // ---- layer 3 (K = 128)
    k_gemm_wmma<HID><<<gemmb, 256, 0, stream>>>(B1, W3, B0, N);
    k_fill<<<nhb, 256, 0, stream>>>(B1, 0.0f, NH);
    k_edge_agg<<<edgeb, 256, 0, stream>>>(ei, B0, dinv, B1, E);
    k_combine_relu<<<nhb, 256, 0, stream>>>(B1, B0, dinv, b3, NH);

    // ---- global mean pool
    k_fill<<<(G * HID + 255) / 256, 256, 0, stream>>>(g, 0.0f, G * HID);
    k_fill<<<(G + 255) / 256, 256, 0, stream>>>(cnt, 0.0f, G);
    k_pool_sum<<<poolb, 256, 0, stream>>>(B1, batch, g, cnt, N);
    k_pool_div<<<(G * HID + 255) / 256, 256, 0, stream>>>(g, cnt, G * HID);

    // ---- heads
    k_heads<<<G, 64, 0, stream>>>(g, Wt1, bt1, Wt2, bt2, Wc1, bc1, Wc2, bc2, out, G);
}
